// AFlowSkipNet_4114578669818
// MI455X (gfx1250) — matrix operand order repaired and checked by reading, not compiled, not run on hardware
//
#include <hip/hip_runtime.h>
#include <hip/hip_bf16.h>

// ---------------------------------------------------------------------------
// AFlowSkipNet on MI455X (gfx1250): conv heads as implicit GEMM on
// v_wmma_f32_16x16x32_bf16, K = t*Cp + c (tap-major, Cp = pow2 >= Cin).
// All intermediate activations are channel-last bf16 [H][W][Cp] so both the
// LDS halo stage and the epilogue are b128 ops. Conv kernel is templated on
// log2(Cp) so the K loop has compile-time trip count (enables pipelining).
// Workspace requirement: ~124 MB (concat buffer dominates).
// ---------------------------------------------------------------------------

typedef __attribute__((ext_vector_type(16))) __bf16 v16bf;
typedef __attribute__((ext_vector_type(8)))  float  v8f;
typedef __attribute__((ext_vector_type(4)))  unsigned int u32x4;

__device__ __forceinline__ unsigned short f2bf(float f) {
  unsigned int u = __float_as_uint(f);
  u += 0x7fffu + ((u >> 16) & 1u);          // round-to-nearest-even
  return (unsigned short)(u >> 16);
}
__device__ __forceinline__ int iclampi(int v, int lo, int hi) {
  return v < lo ? lo : (v > hi ? hi : v);
}

// ---------------------------------------------------------------------------
// Weight pack: fp32 OIHW -> bf16 [Opad][Kpad], K = t*Cp + c  (t = ky*3+kx)
// Zero padding for o >= O and c >= Cin.
// ---------------------------------------------------------------------------
__global__ void pack_weights_kernel(const float* __restrict__ w,
                                    unsigned short* __restrict__ out,
                                    int O, int Cin, int cpshift) {
  int Cp   = 1 << cpshift;
  int Kpad = 9 << cpshift;
  int Opad = ((O + 15) >> 4) << 4;
  int tot  = Opad * Kpad;
  int idx  = blockIdx.x * blockDim.x + threadIdx.x;
  if (idx >= tot) return;
  int o = idx / Kpad;
  int k = idx - o * Kpad;
  int t = k >> cpshift;
  int c = k & (Cp - 1);
  unsigned short v = 0;
  if (o < O && c < Cin) v = f2bf(w[((size_t)o * Cin + c) * 9 + t]);
  out[idx] = v;
}

// ---------------------------------------------------------------------------
// 3x3 conv, stride 1, SAME pad, implicit GEMM on WMMA bf16.
// Input: channel-last bf16 [H][W][Cp] (zero-padded channels).
// Block = 16 consecutive pixels of one row; LDS stages halo tile
// lds[(r*18+xx)*CPP + c], CPP = Cp+8 (breaks pow2 bank strides).
// 8 waves; wave w computes out-channel tile w (16 channels).
// Output: bf16 channel-last [H][W][1<<ocshift]  (or fp32 planes for final).
// ---------------------------------------------------------------------------
#define CONV_TW 16
#define CPPAD   8

template <int CPSHIFT>
__global__ __launch_bounds__(256)
void conv3x3_wmma_kernel(const unsigned short* __restrict__ in,   // bf16 [H][W][Cp]
                         const unsigned short* __restrict__ wpk,  // bf16 [Opad][Kpad]
                         const float* __restrict__ bias,          // [Cout]
                         unsigned short* __restrict__ out_bf,     // bf16 [H][W][Cout] or null
                         float* __restrict__ out_f,               // f32  [Cout][H][W] or null
                         int ocshift, int Cout, int H, int W, int leaky_en) {
  constexpr int Cp  = 1 << CPSHIFT;
  constexpr int CPP = Cp + CPPAD;
  constexpr int Kpad = 9 << CPSHIFT;
  constexpr int CW  = Cp >> 3;                 // 8-element chunks per rx
  __shared__ unsigned short lds_in[54 * CPP];

  const int Opad = ((Cout + 15) >> 4) << 4;
  const int tiles_x = W >> 4;
  const int ty = blockIdx.x / tiles_x;                    // output row
  const int tx = (blockIdx.x - ty * tiles_x) << 4;        // first output col

  // cooperative halo stage: b128 global loads -> b128 LDS stores
  const int nchunks = 54 * CW;
  for (int j = threadIdx.x; j < nchunks; j += blockDim.x) {
    int rx = j >> (CPSHIFT - 3);               // 0..53 = r*18 + xx
    int co = (j & (CW - 1)) << 3;
    int r  = rx / 18;
    int xx = rx - r * 18;
    int y = ty + r - 1;
    int x = tx + xx - 1;
    u32x4 v = {0u, 0u, 0u, 0u};
    if ((unsigned)y < (unsigned)H && (unsigned)x < (unsigned)W)
      v = *(const u32x4*)(in + (((size_t)y * W + x) << CPSHIFT) + co);
    *(u32x4*)(&lds_in[rx * CPP + co]) = v;
  }
  __syncthreads();

  const int lane  = threadIdx.x & 31;
  const int wave  = threadIdx.x >> 5;
  const int px    = lane & 15;                 // B column / D column (pixel)
  const int m     = lane & 15;                 // A row (oc within tile)
  const int ahalf = (lane & 16) ? 8 : 0;       // A K sub-base per lane half
  const int bk    = (lane & 16) ? 8 : 0;       // B K sub-base per lane half

  for (int ot = wave; ot < (Opad >> 4); ot += 8) {
    v8f acc;
#pragma unroll
    for (int r = 0; r < 8; ++r) acc[r] = 0.0f;

    const unsigned short* wrow = wpk + (size_t)(ot * 16 + m) * Kpad + ahalf;

#pragma unroll
    for (int t = 0; t < 9; ++t) {
      const int r  = t / 3;                    // compile-time (t unrolled)
      const int kx = t - r * 3;
      const unsigned short* lrow = &lds_in[(r * 18 + px + kx) * CPP + bk];
      const unsigned short* wt   = wrow + (t << CPSHIFT);
#pragma unroll 4
      for (int cc = 0; cc < Cp; cc += 32) {
        union { v16bf v; u32x4 q[2]; } A, B;
        A.q[0] = *(const u32x4*)(wt + cc);        // k = base+0..7   (16B)
        A.q[1] = *(const u32x4*)(wt + cc + 16);   // k = base+16..23 (16B)
        B.q[0] = *(const u32x4*)(lrow + cc);      // c = cc+bk+0..7
        B.q[1] = *(const u32x4*)(lrow + cc + 16); // c = cc+bk+16..23
        acc = __builtin_amdgcn_wmma_f32_16x16x32_bf16(
            false, A.v, false, B.v, (short)0, acc, false, false);
      }
    }

    // epilogue: D vgpr r -> M = mbase + r (8 consecutive channels), N = px
    const int mbase = ot * 16 + ((lane & 16) ? 8 : 0);
    const size_t pix = (size_t)ty * W + tx + px;
    if (out_bf) {
      union { u32x4 q; unsigned short s[8]; } ov;
#pragma unroll
      for (int r = 0; r < 8; ++r) {
        float v = acc[r] + bias[mbase + r];
        if (leaky_en) v = fmaxf(0.1f * v, v);
        ov.s[r] = f2bf(v);
      }
      *(u32x4*)(out_bf + (pix << ocshift) + mbase) = ov.q;
    } else {
#pragma unroll
      for (int r = 0; r < 8; ++r) {
        int oc = mbase + r;
        if (oc < Cout) {
          float v = acc[r] + bias[oc];
          if (leaky_en) v = fmaxf(0.1f * v, v);
          out_f[(size_t)oc * H * W + pix] = v;
        }
      }
    }
  }
}

// ---------------------------------------------------------------------------
// correlation: leaky( sum_c x[c,h,w]*ypad[c,h+j,w+i] / 256 ), inputs fp32
// plane layout, output channel-last bf16 [H][W][64] zero-padded (49 used).
// ---------------------------------------------------------------------------
__global__ void corr_leaky_kernel(const float* __restrict__ x,
                                  const float* __restrict__ y,
                                  unsigned short* __restrict__ out,
                                  int H, int W) {
  int p = blockIdx.x * blockDim.x + threadIdx.x;
  int HW = H * W;
  if (p >= HW) return;
  int h = p / W, w = p - h * W;

  float acc[49];
#pragma unroll
  for (int o = 0; o < 49; ++o) acc[o] = 0.0f;

  for (int c = 0; c < 256; ++c) {
    float xv = x[(size_t)c * HW + p];
    const float* yc = y + (size_t)c * HW;
#pragma unroll
    for (int j = 0; j < 7; ++j) {
      int yy = h + j - 3;
      if (yy < 0 || yy >= H) continue;
#pragma unroll
      for (int i = 0; i < 7; ++i) {
        int xx = w + i - 3;
        if (xx < 0 || xx >= W) continue;
        acc[j * 7 + i] += xv * yc[yy * W + xx];
      }
    }
  }
  unsigned short* op = out + ((size_t)p << 6);
#pragma unroll
  for (int o = 0; o < 49; ++o) {
    float v = acc[o] * (1.0f / 256.0f);
    v = fmaxf(0.1f * v, v);
    op[o] = f2bf(v);
  }
#pragma unroll
  for (int o = 49; o < 64; ++o) op[o] = 0;
}

// ---------------------------------------------------------------------------
// grid_sample: coords given as 2 planes [gx][gy] (normalized), img fp32
// plane layout. fp32 output: plane layout. bf16 output: channel-last with
// stride `ostride` (concat staging).
// ---------------------------------------------------------------------------
__global__ void grid_sample_kernel(const float* __restrict__ img,
                                   const float* __restrict__ flow, // [2][H][W]
                                   float* __restrict__ outf,
                                   unsigned short* __restrict__ outbf,
                                   int ostride, int C, int H, int W) {
  int p = blockIdx.x * blockDim.x + threadIdx.x;
  int HW = H * W;
  if (p >= HW) return;
  float gx = flow[p];
  float gy = flow[HW + p];
  float ix = fminf(fmaxf(((gx + 1.0f) * W - 1.0f) * 0.5f, 0.0f), (float)(W - 1));
  float iy = fminf(fmaxf(((gy + 1.0f) * H - 1.0f) * 0.5f, 0.0f), (float)(H - 1));
  float x0 = floorf(ix), y0 = floorf(iy);
  float wx = ix - x0, wy = iy - y0;
  int x0i = iclampi((int)x0, 0, W - 1);
  int y0i = iclampi((int)y0, 0, H - 1);
  int x1i = iclampi(x0i + 1, 0, W - 1);
  int y1i = iclampi(y0i + 1, 0, H - 1);
  float w00 = (1.f - wx) * (1.f - wy), w01 = wx * (1.f - wy);
  float w10 = (1.f - wx) * wy,         w11 = wx * wy;
  int i00 = y0i * W + x0i, i01 = y0i * W + x1i;
  int i10 = y1i * W + x0i, i11 = y1i * W + x1i;
  for (int c = 0; c < C; ++c) {
    const float* im = img + (size_t)c * HW;
    float v = im[i00] * w00 + im[i01] * w01 + im[i10] * w10 + im[i11] * w11;
    if (outf) outf[(size_t)c * HW + p] = v;
    else      outbf[(size_t)p * ostride + c] = f2bf(v);
  }
}

// ---------------------------------------------------------------------------
// plane fp32 -> channel-last bf16 transpose (xc into concat second half)
// ---------------------------------------------------------------------------
__global__ void f32_to_bf16_cl_kernel(const float* __restrict__ in, // [C][HW]
                                      unsigned short* __restrict__ out,
                                      int HW, int C, int ostride, int obase) {
  int idx = blockIdx.x * blockDim.x + threadIdx.x;
  int tot = C * HW;
  if (idx >= tot) return;
  int p = idx / C;            // c fastest -> coalesced writes
  int c = idx - p * C;
  out[(size_t)p * ostride + obase + c] = f2bf(in[(size_t)c * HW + p]);
}

// ---------------------------------------------------------------------------
// apply_offset (+ optional grid_sample of last_flow). If lastflow==null the
// normalized grid itself becomes the new flow (level-0 main path).
// ---------------------------------------------------------------------------
__global__ void offset_sample_kernel(const float* __restrict__ off,      // [2][H][W]
                                     const float* __restrict__ lastflow, // [2][H][W] or null
                                     float* __restrict__ outflow,        // [2][H][W]
                                     int H, int W) {
  int p = blockIdx.x * blockDim.x + threadIdx.x;
  int HW = H * W;
  if (p >= HW) return;
  int h = p / W, w = p - h * W;
  float fx = (float)w + off[p];
  float fy = (float)h + off[HW + p];
  float gx = fx / ((W - 1.0f) * 0.5f) - 1.0f;
  float gy = fy / ((H - 1.0f) * 0.5f) - 1.0f;
  if (!lastflow) {
    outflow[p]      = gx;
    outflow[HW + p] = gy;
    return;
  }
  float ix = fminf(fmaxf(((gx + 1.0f) * W - 1.0f) * 0.5f, 0.0f), (float)(W - 1));
  float iy = fminf(fmaxf(((gy + 1.0f) * H - 1.0f) * 0.5f, 0.0f), (float)(H - 1));
  float x0 = floorf(ix), y0 = floorf(iy);
  float wx = ix - x0, wy = iy - y0;
  int x0i = iclampi((int)x0, 0, W - 1);
  int y0i = iclampi((int)y0, 0, H - 1);
  int x1i = iclampi(x0i + 1, 0, W - 1);
  int y1i = iclampi(y0i + 1, 0, H - 1);
  float w00 = (1.f - wx) * (1.f - wy), w01 = wx * (1.f - wy);
  float w10 = (1.f - wx) * wy,         w11 = wx * wy;
  int i00 = y0i * W + x0i, i01 = y0i * W + x1i;
  int i10 = y1i * W + x0i, i11 = y1i * W + x1i;
#pragma unroll
  for (int c = 0; c < 2; ++c) {
    const float* im = lastflow + (size_t)c * HW;
    outflow[(size_t)c * HW + p] =
        im[i00] * w00 + im[i01] * w01 + im[i10] * w10 + im[i11] * w11;
  }
}

// ---------------------------------------------------------------------------
// 2x bilinear upsample (half-pixel centers, edge clamp)
// ---------------------------------------------------------------------------
__global__ void upsample2x_kernel(const float* __restrict__ in,
                                  float* __restrict__ out,
                                  int C, int H, int W) {
  int OH = 2 * H, OW = 2 * W;
  int tot = C * OH * OW;
  int p = blockIdx.x * blockDim.x + threadIdx.x;
  if (p >= tot) return;
  int c   = p / (OH * OW);
  int rem = p - c * (OH * OW);
  int oy = rem / OW, ox = rem - oy * OW;
  float iy = fminf(fmaxf((oy + 0.5f) * 0.5f - 0.5f, 0.0f), (float)(H - 1));
  float ix = fminf(fmaxf((ox + 0.5f) * 0.5f - 0.5f, 0.0f), (float)(W - 1));
  float x0 = floorf(ix), y0 = floorf(iy);
  float wx = ix - x0, wy = iy - y0;
  int x0i = (int)x0, y0i = (int)y0;
  int x1i = iclampi(x0i + 1, 0, W - 1);
  int y1i = iclampi(y0i + 1, 0, H - 1);
  const float* im = in + (size_t)c * H * W;
  float v = im[y0i * W + x0i] * (1.f - wx) * (1.f - wy)
          + im[y0i * W + x1i] * wx * (1.f - wy)
          + im[y1i * W + x0i] * (1.f - wx) * wy
          + im[y1i * W + x1i] * wx * wy;
  out[p] = v;
}

// ---------------------------------------------------------------------------
// Host orchestration
// ---------------------------------------------------------------------------
extern "C" void kernel_launch(void* const* d_in, const int* in_sizes, int n_in,
                              void* d_out, int out_size, void* d_ws, size_t ws_size,
                              hipStream_t stream) {
  (void)in_sizes; (void)n_in; (void)out_size; (void)ws_size;

  // --- inputs: x, xw0,xc0, xw1,xc1, xw2,xc2, xw3,xc3, then params ---
  const float* xw[4]; const float* xc[4];
  for (int k = 0; k < 4; ++k) {
    xw[k] = (const float*)d_in[1 + 2 * k];
    xc[k] = (const float*)d_in[2 + 2 * k];
  }
  // params flattened: main[lvl][layer](w,b) x4x4, then refine[lvl][layer](w,b)
  auto mW = [&](int lvl, int l) { return (const float*)d_in[9  + (lvl * 4 + l) * 2]; };
  auto mB = [&](int lvl, int l) { return (const float*)d_in[9  + (lvl * 4 + l) * 2 + 1]; };
  auto rW = [&](int lvl, int l) { return (const float*)d_in[41 + (lvl * 4 + l) * 2]; };
  auto rB = [&](int lvl, int l) { return (const float*)d_in[41 + (lvl * 4 + l) * 2 + 1]; };

  static const int main_ci[4]  = {49, 128, 64, 32}, main_co[4] = {128, 64, 32, 2};
  static const int ref_ci[4]   = {512, 128, 64, 32}, ref_co[4] = {128, 64, 32, 2};
  static const int main_cps[4] = {6, 7, 6, 5};   // log2(Cp) per layer (input)
  static const int ref_cps[4]  = {9, 7, 6, 5};
  static const int ocs[4]      = {7, 6, 5, 0};   // log2(Cout) for bf16 outputs

  // --- packed-weight offsets (elements) ---
  size_t wpk_off[4][2][4];
  size_t wtot = 0;
  for (int lvl = 0; lvl < 4; ++lvl)
    for (int head = 0; head < 2; ++head)
      for (int l = 0; l < 4; ++l) {
        int co = head ? ref_co[l] : main_co[l];
        int cs = head ? ref_cps[l] : main_cps[l];
        size_t Kp = (size_t)9 << cs;
        size_t Op = (size_t)(((co + 15) / 16) * 16);
        wpk_off[lvl][head][l] = wtot;
        wtot += Op * Kp;
      }

  // --- carve workspace ---
  char* wsp = (char*)d_ws;
  auto carve = [&](size_t bytes) -> char* {
    char* p = wsp;
    wsp += (bytes + 255) & ~(size_t)255;
    return p;
  };
  unsigned short* WPK    = (unsigned short*)carve(wtot * 2);
  float*          XWA    = (float*)carve((size_t)256 * 128 * 128 * 4);          // warped xw (main, fp32 planes)
  unsigned short* CORR   = (unsigned short*)carve((size_t)64 * 128 * 128 * 2);  // corr bf16 [HW][64]
  unsigned short* CONCAT = (unsigned short*)carve((size_t)512 * 256 * 256 * 2); // [HW][512]
  unsigned short* H1     = (unsigned short*)carve((size_t)128 * 256 * 256 * 2); // [HW][128]
  unsigned short* H2     = (unsigned short*)carve((size_t)64  * 256 * 256 * 2); // [HW][64]
  unsigned short* H3     = (unsigned short*)carve((size_t)32  * 256 * 256 * 2); // [HW][32]
  float*          FCONV  = (float*)carve((size_t)2 * 256 * 256 * 4);            // conv-head flow out
  float*          FA     = (float*)carve((size_t)2 * 256 * 256 * 4);            // flow ping
  float*          FB     = (float*)carve((size_t)2 * 256 * 256 * 4);            // flow pong

  // --- pack all weights to bf16 [Opad][Kpad], K = t*Cp + c ---
  for (int lvl = 0; lvl < 4; ++lvl)
    for (int head = 0; head < 2; ++head)
      for (int l = 0; l < 4; ++l) {
        int ci = head ? ref_ci[l] : main_ci[l];
        int co = head ? ref_co[l] : main_co[l];
        int cs = head ? ref_cps[l] : main_cps[l];
        int Kp = 9 << cs;
        int Op = ((co + 15) / 16) * 16;
        int tot = Op * Kp;
        const float* w = head ? rW(lvl, l) : mW(lvl, l);
        pack_weights_kernel<<<(tot + 255) / 256, 256, 0, stream>>>(
            w, WPK + wpk_off[lvl][head][l], co, ci, cs);
      }

  // --- templated conv dispatch ---
  auto launch_conv = [&](int cps, dim3 grid, const unsigned short* cur,
                         const unsigned short* wp, const float* b,
                         unsigned short* ob, float* of, int ocshift, int Cout,
                         int s, int leaky) {
    switch (cps) {
      case 5: conv3x3_wmma_kernel<5><<<grid, 256, 0, stream>>>(cur, wp, b, ob, of, ocshift, Cout, s, s, leaky); break;
      case 6: conv3x3_wmma_kernel<6><<<grid, 256, 0, stream>>>(cur, wp, b, ob, of, ocshift, Cout, s, s, leaky); break;
      case 7: conv3x3_wmma_kernel<7><<<grid, 256, 0, stream>>>(cur, wp, b, ob, of, ocshift, Cout, s, s, leaky); break;
      default: conv3x3_wmma_kernel<9><<<grid, 256, 0, stream>>>(cur, wp, b, ob, of, ocshift, Cout, s, s, leaky); break;
    }
  };

  // --- conv head runner ---
  auto run_head = [&](const unsigned short* in0, int lvl, int head, int s,
                      float* flow_out) {
    const unsigned short* cur = in0;
    const int* co  = head ? ref_co  : main_co;
    const int* cps = head ? ref_cps : main_cps;
    for (int l = 0; l < 4; ++l) {
      bool last = (l == 3);
      unsigned short* ob = last ? nullptr : (l == 0 ? H1 : (l == 1 ? H2 : H3));
      float* of = last ? flow_out : nullptr;
      const float* b = head ? rB(lvl, l) : mB(lvl, l);
      dim3 grid(s * (s / CONV_TW));
      launch_conv(cps[l], grid, cur, WPK + wpk_off[lvl][head][l], b, ob, of,
                  ocs[l], co[l], s, last ? 0 : 1);
      cur = ob;
    }
  };

  // --- pyramid loop (coarse 32 -> fine 256) ---
  float* lf = nullptr;   // last_flow (normalized grid, [2][s][s])
  for (int i = 0; i < 4; ++i) {
    int s  = 32 << i;
    int HW = s * s;
    const float* xwi = xw[3 - i];
    const float* xci = xc[3 - i];
    int pix_blocks = (HW + 255) / 256;

    if (i < 3) {
      const float* xwa = xwi;
      if (lf) {
        grid_sample_kernel<<<pix_blocks, 256, 0, stream>>>(
            xwi, lf, XWA, nullptr, 0, 256, s, s);
        xwa = XWA;
      }
      corr_leaky_kernel<<<pix_blocks, 256, 0, stream>>>(xwa, xci, CORR, s, s);
      run_head(CORR, i, /*head=*/0, s, FCONV);
      float* nf = (lf == FA) ? FB : FA;
      offset_sample_kernel<<<pix_blocks, 256, 0, stream>>>(FCONV, lf, nf, s, s);
      lf = nf;
    }

    // refine: concat channel-last [HW][512] = [warped xw bf16 | xc bf16]
    grid_sample_kernel<<<pix_blocks, 256, 0, stream>>>(
        xwi, lf, nullptr, CONCAT, 512, 256, s, s);
    f32_to_bf16_cl_kernel<<<(256 * HW + 255) / 256, 256, 0, stream>>>(
        xci, CONCAT, HW, 256, 512, 256);
    run_head(CONCAT, i, /*head=*/1, s, FCONV);

    float* nf = (lf == FA) ? FB : FA;
    offset_sample_kernel<<<pix_blocks, 256, 0, stream>>>(FCONV, lf, nf, s, s);

    int up_tot = 2 * (2 * s) * (2 * s);
    if (i == 3) {
      upsample2x_kernel<<<(up_tot + 255) / 256, 256, 0, stream>>>(
          nf, (float*)d_out, 2, s, s);
    } else {
      float* uf = (nf == FA) ? FB : FA;
      upsample2x_kernel<<<(up_tot + 255) / 256, 256, 0, stream>>>(
          nf, uf, 2, s, s);
      lf = uf;
    }
  }
}
